// LSTMGCN_56083682951833
// MI455X (gfx1250) — matrix operand order repaired and checked by reading, not compiled, not run on hardware
//
#include <hip/hip_runtime.h>
#include <hip/hip_bf16.h>

// ---------------------------------------------------------------------------
// LSTM (T=200, H=50, in=2) -> GCNConv(50->16) -> ReLU -> GCNConv(16->1)
// -> softmax over a length-1 axis (== exp(z-z) == 1, NaN-propagating).
// LSTM is compute bound (~2.1e11 FLOP vs 80MB input) -> v_wmma_f32_16x16x32_f16.
// Bias is folded into the GEMM via a constant-1 K row; sigmoid uses v_rcp_f32.
// ---------------------------------------------------------------------------

typedef __attribute__((ext_vector_type(16))) _Float16 v16h;
typedef __attribute__((ext_vector_type(8)))  float    v8f;

#define LSTM_T   200
#define LSTM_H   50
#define KPAD     64        // K = [h(50) | x(2) | 1 (bias) | zero pad] -> 64
#define WAVES    4
#define TILE_M   16
#define BLK_SMP  (WAVES * TILE_M)

union AF { uint4 u[2]; v16h h; };   // 32B: one 16x32 / 32x16 f16 WMMA fragment

__device__ __forceinline__ float sigf(float x) {
    // 1/(1+exp(-x)) with v_rcp_f32 instead of IEEE division
    return __builtin_amdgcn_rcpf(1.0f + __expf(-x));
}
__device__ __forceinline__ float tanhf_(float x) {
    return __builtin_fmaf(2.0f, sigf(2.0f * x), -1.0f);
}

__device__ __forceinline__ unsigned pack2h(float a, float b) {
    union { _Float16 h[2]; unsigned u; } p;
    p.h[0] = (_Float16)a; p.h[1] = (_Float16)b;
    return p.u;
}

// B[k, col] of the fused weight matrix, col = g*64 + u (u padded 50->64).
// K rows: 0..49 = W_hh, 50..51 = W_ih, 52 = fused bias (A supplies constant 1).
__device__ __forceinline__ float wval(const float* W_hh, const float* W_ih,
                                      const float* b_ih, const float* b_hh,
                                      int g, int u, int k) {
    if (u >= LSTM_H) return 0.0f;
    const int row = g * LSTM_H + u;                 // row of W_ih/W_hh [200, *]
    if (k < LSTM_H)  return W_hh[row * LSTM_H + k]; // recurrent part
    if (k == LSTM_H)     return W_ih[row * 2 + 0];  // input x0
    if (k == LSTM_H + 1) return W_ih[row * 2 + 1];  // input x1
    if (k == LSTM_H + 2) return b_ih[row] + b_hh[row]; // bias row
    return 0.0f;                                    // K padding
}

// ---------------------------------------------------------------------------
// LSTM kernel: 128 threads = 4 waves, each wave owns a 16-sample tile.
// Per step per wave: 4 col-groups x 4 gates x 2 k-steps = 32 wmma_f32_16x16x32_f16.
// ---------------------------------------------------------------------------
__global__ __launch_bounds__(128)
void lstm_wmma_kernel(const float* __restrict__ data,    // [T, N, 2]
                      const float* __restrict__ W_ih,    // [200, 2]
                      const float* __restrict__ W_hh,    // [200, 50]
                      const float* __restrict__ b_ih,    // [200]
                      const float* __restrict__ b_hh,    // [200]
                      float* __restrict__ hout,          // [N, 50]
                      int N) {
    // Fragment-packed B in LDS: frag = tile*2 + kk, per lane 32B (2 x b128).
    __shared__ uint4 bfrag[32][32][2];                           // 32 KB
    __shared__ __align__(16) _Float16 hbuf[WAVES][TILE_M][KPAD]; // 8 KB, wave-private

    const int wave = threadIdx.x >> 5;
    const int lane = threadIdx.x & 31;
    const int n    = lane & 15;            // C/B column within a 16-wide tile
    const int base = blockIdx.x * BLK_SMP + wave * TILE_M;

    // ---- build fragment-packed weights (once per block) ----
    for (int p = threadIdx.x; p < 32 * 32; p += blockDim.x) {
        const int frag = p >> 5, ln = p & 31;
        const int tile = frag >> 1, kk = frag & 1;
        const int col  = tile * 16 + (ln & 15);   // 0..255
        const int g    = col >> 6;                // gate
        const int u    = col & 63;                // unit within gate (padded)
        const int kb   = ((ln < 16) ? 0 : 8) + 32 * kk;
        unsigned d[8];
#pragma unroll
        for (int j = 0; j < 8; ++j) {
            const int k0 = kb + 2 * j + ((j >= 4) ? 8 : 0);
            d[j] = pack2h(wval(W_hh, W_ih, b_ih, b_hh, g, u, k0),
                          wval(W_hh, W_ih, b_ih, b_hh, g, u, k0 + 1));
        }
        bfrag[frag][ln][0] = make_uint4(d[0], d[1], d[2], d[3]);
        bfrag[frag][ln][1] = make_uint4(d[4], d[5], d[6], d[7]);
    }

    // ---- zero own wave's h/x staging buffer; set bias-row constant 1 ----
    _Float16* hb = &hbuf[wave][0][0];
    for (int i = lane; i < TILE_M * KPAD; i += 32) hb[i] = (_Float16)0.0f;
    if (lane < TILE_M) hbuf[wave][lane][LSTM_H + 2] = (_Float16)1.0f;

    __syncthreads();   // bfrag visible to all waves

    v8f cst[4] = {};   // cell state, same layout as C fragments

    for (int t = 0; t < LSTM_T; ++t) {
        // stage x_t into K slots 50..51 (lanes 0..15, one row each)
        if (lane < TILE_M) {
            int s = base + lane; if (s >= N) s = N - 1;
            const float* xp = data + ((size_t)t * N + s) * 2;
            hbuf[wave][lane][LSTM_H]     = (_Float16)xp[0];
            hbuf[wave][lane][LSTM_H + 1] = (_Float16)xp[1];
            if (t + 1 < LSTM_T)
                __builtin_prefetch(data + ((size_t)(t + 1) * N + s) * 2, 0, 1);
        }

        // A fragments: 16x32 f16 per-lane layout, two b128 loads each
        AF a0, a1;
        {
            const int m  = n;
            const int kb = (lane < 16) ? 0 : 8;
            a0.u[0] = *(const uint4*)&hbuf[wave][m][kb];
            a0.u[1] = *(const uint4*)&hbuf[wave][m][kb + 16];
            a1.u[0] = *(const uint4*)&hbuf[wave][m][kb + 32];
            a1.u[1] = *(const uint4*)&hbuf[wave][m][kb + 48];
        }

#pragma unroll
        for (int tg = 0; tg < 4; ++tg) {      // 16-column group within each gate
            // Scheduling barrier: keep each tg's B-fragment LDS loads inside
            // its own group so the compiler cannot hoist all 32 fragments into
            // registers (which pushes VGPRs >256 and triggers s_set_vgpr_msb
            // on nearly every instruction).
            asm volatile("" ::: "memory");
            v8f z[4];
#pragma unroll
            for (int g = 0; g < 4; ++g) {     // gate i/f/g/o -> tiles tg, tg+4, ...
                const int tile = g * 4 + tg;
                AF b0, b1;
                b0.u[0] = bfrag[tile * 2 + 0][lane][0];
                b0.u[1] = bfrag[tile * 2 + 0][lane][1];
                b1.u[0] = bfrag[tile * 2 + 1][lane][0];
                b1.u[1] = bfrag[tile * 2 + 1][lane][1];
                v8f c = {};
                c = __builtin_amdgcn_wmma_f32_16x16x32_f16(false, a0.h, false, b0.h,
                                                           (short)0, c, false, false);
                c = __builtin_amdgcn_wmma_f32_16x16x32_f16(false, a1.h, false, b1.h,
                                                           (short)0, c, false, false);
                z[g] = c;   // z already includes bias via the K=52 row
            }
            // elementwise gate math; identical C-fragment coords across gates
#pragma unroll
            for (int r = 0; r < 8; ++r) {
                const float cc = sigf(z[1][r]) * cst[tg][r]
                               + sigf(z[0][r]) * tanhf_(z[2][r]);
                cst[tg][r]     = cc;
                const float hh = sigf(z[3][r]) * tanhf_(cc);
                // u = tg*16 + n; only tg==3 can reach the padded range
                if (tg < 3 || n < 2) {
                    const int m = r + ((lane < 16) ? 0 : 8);
                    hbuf[wave][m][tg * 16 + n] = (_Float16)hh; // never touches x/bias slots
                }
            }
        }
    }

    // write final hidden state
    for (int i = lane; i < TILE_M * LSTM_H; i += 32) {
        const int m = i / LSTM_H, u = i % LSTM_H;
        const int s = base + m;
        if (s < N) hout[(size_t)s * LSTM_H + u] = (float)hbuf[wave][m][u];
    }
}

// ---------------------------------------------------------------------------
// GCN stages (memory/atomic bound; accumulators are L2-resident at 3.2MB)
// ---------------------------------------------------------------------------
__global__ void deg_init_k(float* deg, int N) {
    int i = blockIdx.x * blockDim.x + threadIdx.x;
    if (i < N) deg[i] = 1.0f;                        // self loop
}
__global__ void deg_edges_k(const long long* __restrict__ ei, float* deg, int E) {
    int e = blockIdx.x * blockDim.x + threadIdx.x;
    if (e < E) atomicAdd(&deg[(int)ei[(size_t)E + e]], 1.0f);
}
__global__ void rsqrt_k(const float* deg, float* dinv, int N) {
    int i = blockIdx.x * blockDim.x + threadIdx.x;
    if (i < N) dinv[i] = rsqrtf(deg[i]);             // deg >= 1 always
}
__global__ void xw1_k(const float* __restrict__ h, const float* __restrict__ W1,
                      float* __restrict__ xw, int N) {
    int nId = blockIdx.x * blockDim.x + threadIdx.x;
    if (nId >= N) return;
    const float* hr = h + (size_t)nId * LSTM_H;
    float acc[16];
#pragma unroll
    for (int j = 0; j < 16; ++j) acc[j] = 0.0f;
    for (int k = 0; k < LSTM_H; ++k) {
        const float hv = hr[k];
#pragma unroll
        for (int j = 0; j < 16; ++j) acc[j] += hv * W1[k * 16 + j];
    }
#pragma unroll
    for (int j = 0; j < 16; ++j) xw[(size_t)nId * 16 + j] = acc[j];
}
__global__ void scatter1_k(const long long* __restrict__ ei,
                           const float* __restrict__ xw,
                           const float* __restrict__ dinv,
                           float* __restrict__ agg, int E) {
    long long tid = (long long)blockIdx.x * blockDim.x + threadIdx.x;
    int e = (int)(tid >> 4), j = (int)(tid & 15);
    if (e >= E) return;
    const int r = (int)ei[e], c = (int)ei[(size_t)E + e];
    atomicAdd(&agg[(size_t)c * 16 + j], xw[(size_t)r * 16 + j] * dinv[r] * dinv[c]);
}
__global__ void relu1_k(const float* __restrict__ agg, const float* __restrict__ xw,
                        const float* __restrict__ dinv, const float* __restrict__ b1,
                        float* __restrict__ out1, int N) {
    long long tid = (long long)blockIdx.x * blockDim.x + threadIdx.x;
    if (tid >= (long long)N * 16) return;
    const int nId = (int)(tid >> 4), j = (int)(tid & 15);
    const float v = agg[tid] + xw[tid] * dinv[nId] * dinv[nId] + b1[j];
    out1[tid] = v > 0.0f ? v : 0.0f;
}
__global__ void xw2_k(const float* __restrict__ out1, const float* __restrict__ W2,
                      float* __restrict__ x2, int N) {
    int nId = blockIdx.x * blockDim.x + threadIdx.x;
    if (nId >= N) return;
    float a = 0.0f;
#pragma unroll
    for (int j = 0; j < 16; ++j) a += out1[(size_t)nId * 16 + j] * W2[j];
    x2[nId] = a;
}
__global__ void scatter2_k(const long long* __restrict__ ei,
                           const float* __restrict__ x2,
                           const float* __restrict__ dinv,
                           float* __restrict__ agg2, int E) {
    int e = blockIdx.x * blockDim.x + threadIdx.x;
    if (e >= E) return;
    const int r = (int)ei[e], c = (int)ei[(size_t)E + e];
    atomicAdd(&agg2[c], x2[r] * dinv[r] * dinv[c]);
}
__global__ void final_k(const float* __restrict__ agg2, const float* __restrict__ x2,
                        const float* __restrict__ dinv, const float* __restrict__ b2,
                        float* __restrict__ out, int N) {
    int nId = blockIdx.x * blockDim.x + threadIdx.x;
    if (nId >= N) return;
    const float z = agg2[nId] + x2[nId] * dinv[nId] * dinv[nId] + b2[0];
    out[nId] = __expf(z - z);   // softmax over axis of length 1: exactly 1 (NaN-propagating)
}

// ---------------------------------------------------------------------------
extern "C" void kernel_launch(void* const* d_in, const int* in_sizes, int n_in,
                              void* d_out, int out_size, void* d_ws, size_t ws_size,
                              hipStream_t stream) {
    const float*     data = (const float*)d_in[0];      // [T, N, 2]
    const long long* ei   = (const long long*)d_in[1];  // [2, E] int64
    const float*     W_ih = (const float*)d_in[2];
    const float*     W_hh = (const float*)d_in[3];
    const float*     b_ih = (const float*)d_in[4];
    const float*     b_hh = (const float*)d_in[5];
    const float*     W1   = (const float*)d_in[6];
    const float*     b1   = (const float*)d_in[7];
    const float*     W2   = (const float*)d_in[8];
    const float*     b2   = (const float*)d_in[9];
    float* out = (float*)d_out;

    const int N = out_size;               // output is [N, 1]
    const int E = in_sizes[1] / 2;

    // workspace carve-up (all f32): N*(50+16+16+16+1+1+1+1) = 102*N floats
    float* ws    = (float*)d_ws;
    float* h_T   = ws;                       ws += (size_t)N * LSTM_H;
    float* xw1   = ws;                       ws += (size_t)N * 16;
    float* agg1  = ws;                       ws += (size_t)N * 16;
    float* out1  = ws;                       ws += (size_t)N * 16;
    float* x2    = ws;                       ws += N;
    float* agg2  = ws;                       ws += N;
    float* deg   = ws;                       ws += N;
    float* dinv  = ws;                       ws += N;
    (void)ws_size; (void)n_in;

    (void)hipMemsetAsync(agg1, 0, (size_t)N * 16 * sizeof(float), stream);
    (void)hipMemsetAsync(agg2, 0, (size_t)N * sizeof(float), stream);

    // LSTM: 4 waves/block, 64 samples/block
    const int nblk = (N + BLK_SMP - 1) / BLK_SMP;
    lstm_wmma_kernel<<<nblk, 128, 0, stream>>>(data, W_ih, W_hh, b_ih, b_hh, h_T, N);

    const int t256 = 256;
    deg_init_k <<<(N + t256 - 1) / t256, t256, 0, stream>>>(deg, N);
    deg_edges_k<<<(E + t256 - 1) / t256, t256, 0, stream>>>(ei, deg, E);
    rsqrt_k    <<<(N + t256 - 1) / t256, t256, 0, stream>>>(deg, dinv, N);

    xw1_k      <<<(N + 127) / 128, 128, 0, stream>>>(h_T, W1, xw1, N);
    {
        long long tot = (long long)E * 16;
        scatter1_k<<<(unsigned)((tot + t256 - 1) / t256), t256, 0, stream>>>(ei, xw1, dinv, agg1, E);
    }
    {
        long long tot = (long long)N * 16;
        relu1_k<<<(unsigned)((tot + t256 - 1) / t256), t256, 0, stream>>>(agg1, xw1, dinv, b1, out1, N);
    }
    xw2_k      <<<(N + t256 - 1) / t256, t256, 0, stream>>>(out1, W2, x2, N);
    scatter2_k <<<(E + t256 - 1) / t256, t256, 0, stream>>>(ei, x2, dinv, agg2, E);
    final_k    <<<(N + t256 - 1) / t256, t256, 0, stream>>>(agg2, x2, dinv, b2, out, N);
}
